// Net_60103772340433
// MI455X (gfx1250) — compile-verified, matrix operand order checked by reading, not run on hardware
//
#include <hip/hip_runtime.h>
#include <math.h>

// ---------------------------------------------------------------------------
// Point Transformer forward for gfx1250 (MI455X).
// All dense linear algebra goes through one WMMA GEMM kernel:
//   v_wmma_f32_16x16x32_f16  (f16 operands, f32 accumulate)
// Bandwidth/latency bound workload (~8 GFLOP, ~100s MB @ 23.3 TB/s HBM), so
// the GEMM data path is branchless: clamped addresses (garbage lanes discarded
// at the guarded store), K-tail zeros via v_cndmask, float4 A loads, and a
// compile-time N so all 16 strided B loads are one clause of immediate-offset
// global_load_b32 off a single advancing pointer (no per-element 64-bit muls).
// ---------------------------------------------------------------------------

#define WKNN 16

typedef __attribute__((ext_vector_type(16))) _Float16 v16h;
typedef __attribute__((ext_vector_type(8)))  float    v8f;

// ------------------------- WMMA GEMM -------------------------------------
// C[M,N] = act( A[M,Kd] @ W[Kd,N] + bias ),  A,W row-major f32 in memory.
// One wave computes one 16x16 tile; 2 waves/block share the A tile.
// Fragment layouts per CDNA5 ISA 7.12.2:
//   A (16x32 f16): lane<16 holds row M=lane, K = {0..7,16..23};
//                  lane>=16 holds row M=lane-16, K = {8..15,24..31}
//   B (32x16 f16): mirrored (per-column striping, column N = lane&15)
//   C/D (16x16 f32): vgpr r, lane<16 -> (M=r, N=lane); lane>=16 -> (M=r+8)
// NCT > 0: compile-time N (immediate-offset B loads / C stores). NCT==0: runtime N.
template <int NCT, bool KALIGNED>
__global__ __launch_bounds__(64) void gemm_wmma_kernel(
    const float* __restrict__ A, const float* __restrict__ W,
    const float* __restrict__ bias, float* __restrict__ Cm,
    int M, int Kd, int Nrt, int relu)
{
  const int N = (NCT > 0) ? NCT : Nrt;
  const int lane = threadIdx.x & 31;
  const int wave = threadIdx.x >> 5;
  const int tn   = (blockIdx.x * 2 + wave) << 4;
  const int tm   = blockIdx.y << 4;
  const int hsel = lane >> 4;     // which K-half this lane covers
  const int sub  = lane & 15;
  // Clamped addresses: OOB rows/cols compute garbage that is never stored.
  const int arow = (tm + sub < M) ? (tm + sub) : (M - 1);
  const int bcol = (tn + sub < N) ? (tn + sub) : (N - 1);
  // hsel folded into base pointers: fragment offsets are {0..7, 16..23}.
  const float* __restrict__ Ap = A + (size_t)arow * Kd + hsel * 8;
  const float* __restrict__ Wp = W + (size_t)(hsel * 8) * N + bcol;

  v8f acc = {};
  if (KALIGNED) {
    // Kd % 32 == 0: fully unconditional, vectorized A loads, immediate-offset
    // B loads off a single advancing pointer.
#pragma unroll 2
    for (int k0 = 0; k0 < Kd; k0 += 32) {
      const float4 a0 = *(const float4*)(Ap + 0);
      const float4 a1 = *(const float4*)(Ap + 4);
      const float4 a2 = *(const float4*)(Ap + 16);
      const float4 a3 = *(const float4*)(Ap + 20);
      v16h a, b;
      a[0]  = (_Float16)a0.x; a[1]  = (_Float16)a0.y;
      a[2]  = (_Float16)a0.z; a[3]  = (_Float16)a0.w;
      a[4]  = (_Float16)a1.x; a[5]  = (_Float16)a1.y;
      a[6]  = (_Float16)a1.z; a[7]  = (_Float16)a1.w;
      a[8]  = (_Float16)a2.x; a[9]  = (_Float16)a2.y;
      a[10] = (_Float16)a2.z; a[11] = (_Float16)a2.w;
      a[12] = (_Float16)a3.x; a[13] = (_Float16)a3.y;
      a[14] = (_Float16)a3.z; a[15] = (_Float16)a3.w;
#pragma unroll
      for (int j = 0; j < 8; ++j)
        b[j] = (_Float16)Wp[(size_t)j * N];            // offsets 0..7   (xN)
#pragma unroll
      for (int j = 0; j < 8; ++j)
        b[j + 8] = (_Float16)Wp[(size_t)(16 + j) * N]; // offsets 16..23 (xN)
      acc = __builtin_amdgcn_wmma_f32_16x16x32_f16(false, a, false, b,
                                                   (short)0, acc, false, false);
      Ap += 32;
      Wp += (size_t)32 * N;
    }
  } else {
    // Generic Kd (3, 6): unconditional clamped loads + branchless zero-select.
    for (int k0 = 0; k0 < Kd; k0 += 32) {
      v16h a, b;
#pragma unroll
      for (int j = 0; j < 16; ++j) {
        const int off = (j < 8) ? j : (j + 8);         // {0..7, 16..23}
        const int kk = k0 + hsel * 8 + off;
        const int kc = (kk < Kd) ? kk : (Kd - 1);
        const float av = A[(size_t)arow * Kd + kc];
        const float bv = W[(size_t)kc * N + bcol];
        const bool ok = kk < Kd;
        a[j] = (_Float16)(ok ? av : 0.0f);
        b[j] = (_Float16)(ok ? bv : 0.0f);
      }
      acc = __builtin_amdgcn_wmma_f32_16x16x32_f16(false, a, false, b,
                                                   (short)0, acc, false, false);
    }
  }

  const int ocol = tn + sub;
  if (ocol < N) {
    const float bv = bias ? bias[ocol] : 0.0f;
    const int obase = tm + (hsel << 3);
    float* __restrict__ op = Cm + (size_t)obase * N + ocol;
#pragma unroll
    for (int r = 0; r < 8; ++r) {
      if (obase + r < M) {
        float vv = acc[r] + bv;
        if (relu) vv = fmaxf(vv, 0.0f);
        op[(size_t)r * N] = vv;
      }
    }
  }
}

// ------------------------- KNN (brute force, top-K by insertion) ----------
template <int KN>
__global__ void knn_kernel(const float* __restrict__ base, int nb,
                           const float* __restrict__ query, int nq,
                           int* __restrict__ nbr, float* __restrict__ nd2)
{
  int q = blockIdx.x * blockDim.x + threadIdx.x;
  if (q >= nq) return;
  const float qx = query[q * 3 + 0];
  const float qy = query[q * 3 + 1];
  const float qz = query[q * 3 + 2];
  float bestd[KN];
  int   besti[KN];
#pragma unroll
  for (int j = 0; j < KN; ++j) { bestd[j] = 3.4e38f; besti[j] = 0; }
  for (int p = 0; p < nb; ++p) {
    const float dx = qx - base[p * 3 + 0];
    const float dy = qy - base[p * 3 + 1];
    const float dz = qz - base[p * 3 + 2];
    const float d = dx * dx + dy * dy + dz * dz;
    if (d < bestd[KN - 1]) {   // strict < keeps earlier index on ties (top_k)
      int j = KN - 1;
      while (j > 0 && bestd[j - 1] > d) {
        bestd[j] = bestd[j - 1]; besti[j] = besti[j - 1]; --j;
      }
      bestd[j] = d; besti[j] = p;
    }
  }
#pragma unroll
  for (int j = 0; j < KN; ++j) {
    nbr[(size_t)q * KN + j] = besti[j];
    if (nd2) nd2[(size_t)q * KN + j] = bestd[j];
  }
}

// ------------------------- FPS (single block, sequential argmax) ----------
__global__ void fps_kernel(const float* __restrict__ pos, int n, int m,
                           int* __restrict__ idx, float* __restrict__ dist)
{
  __shared__ float sval[512];
  __shared__ int   sidx[512];
  const int tid = threadIdx.x;
  const int nt  = blockDim.x;
  const float p0x = pos[0], p0y = pos[1], p0z = pos[2];
  for (int i = tid; i < n; i += nt) {
    const float dx = pos[i * 3 + 0] - p0x;
    const float dy = pos[i * 3 + 1] - p0y;
    const float dz = pos[i * 3 + 2] - p0z;
    dist[i] = dx * dx + dy * dy + dz * dz;
  }
  if (tid == 0) idx[0] = 0;
  __syncthreads();
  for (int it = 1; it < m; ++it) {
    float bv = -1.0f; int bi = 0x7fffffff;
    for (int i = tid; i < n; i += nt) {
      const float v = dist[i];
      if (v > bv || (v == bv && i < bi)) { bv = v; bi = i; }
    }
    sval[tid] = bv; sidx[tid] = bi;
    __syncthreads();
    for (int s = nt >> 1; s > 0; s >>= 1) {
      if (tid < s) {
        const float ov = sval[tid + s]; const int oi = sidx[tid + s];
        if (ov > sval[tid] || (ov == sval[tid] && oi < sidx[tid])) {
          sval[tid] = ov; sidx[tid] = oi;
        }
      }
      __syncthreads();
    }
    const int nxt = sidx[0];
    if (tid == 0) idx[it] = nxt;
    const float nx = pos[nxt * 3 + 0];
    const float ny = pos[nxt * 3 + 1];
    const float nz = pos[nxt * 3 + 2];
    for (int i = tid; i < n; i += nt) {
      const float dx = pos[i * 3 + 0] - nx;
      const float dy = pos[i * 3 + 1] - ny;
      const float dz = pos[i * 3 + 2] - nz;
      const float d = dx * dx + dy * dy + dz * dz;
      if (d < dist[i]) dist[i] = d;
    }
    __syncthreads();
  }
}

// ------------------------- small elementwise kernels ----------------------
__global__ void cat_pos_x_kernel(const float* __restrict__ pos,
                                 const float* __restrict__ x,
                                 float* __restrict__ out, int n)
{
  int i = blockIdx.x * blockDim.x + threadIdx.x;
  if (i >= n) return;
  out[i * 6 + 0] = pos[i * 3 + 0];
  out[i * 6 + 1] = pos[i * 3 + 1];
  out[i * 6 + 2] = pos[i * 3 + 2];
  out[i * 6 + 3] = x[i * 3 + 0];
  out[i * 6 + 4] = x[i * 3 + 1];
  out[i * 6 + 5] = x[i * 3 + 2];
}

__global__ void fill_dst_kernel(int* __restrict__ dst, int E, int Kn)
{
  int e = blockIdx.x * blockDim.x + threadIdx.x;
  if (e < E) dst[e] = e / Kn;
}

__global__ void edge_posdiff_kernel(const float* __restrict__ pos,
                                    const int* __restrict__ src,
                                    const int* __restrict__ dst,
                                    float* __restrict__ out, int E)
{
  int e = blockIdx.x * blockDim.x + threadIdx.x;
  if (e >= E) return;
  const int s = src[e], d = dst[e];
  out[(size_t)e * 3 + 0] = pos[d * 3 + 0] - pos[s * 3 + 0];
  out[(size_t)e * 3 + 1] = pos[d * 3 + 1] - pos[s * 3 + 1];
  out[(size_t)e * 3 + 2] = pos[d * 3 + 2] - pos[s * 3 + 2];
}

__global__ void edge_alpha_pre_kernel(const float* __restrict__ adst,
                                      const float* __restrict__ asrc,
                                      const float* __restrict__ delta,
                                      const int* __restrict__ src,
                                      const int* __restrict__ dst,
                                      float* __restrict__ out, int E, int C)
{
  int t = blockIdx.x * blockDim.x + threadIdx.x;
  if (t >= E * C) return;
  const int e = t / C, c = t - e * C;
  out[t] = adst[(size_t)dst[e] * C + c] - asrc[(size_t)src[e] * C + c] + delta[t];
}

// per-destination channelwise softmax over K consecutive edges + weighted sum
__global__ void tb_aggregate_kernel(const float* __restrict__ alpha,
                                    const float* __restrict__ vfeat,
                                    const float* __restrict__ delta,
                                    const int* __restrict__ src,
                                    float* __restrict__ out, int n, int C, int Kn)
{
  int t = blockIdx.x * blockDim.x + threadIdx.x;
  if (t >= n * C) return;
  const int i = t / C, c = t - i * C;
  const float* arow = alpha + (size_t)i * Kn * C + c;
  float m = -3.4e38f;
  for (int j = 0; j < Kn; ++j) m = fmaxf(m, arow[(size_t)j * C]);
  float s = 0.0f, acc = 0.0f;
  for (int j = 0; j < Kn; ++j) {
    const float e = expf(arow[(size_t)j * C] - m);
    s += e;
    const int sj = src[i * Kn + j];
    acc += e * (vfeat[(size_t)sj * C + c] + delta[((size_t)i * Kn + j) * C + c]);
  }
  out[(size_t)i * C + c] = acc / s;
}

__global__ void gather_pos_kernel(const float* __restrict__ pos,
                                  const int* __restrict__ idx,
                                  float* __restrict__ out, int m)
{
  int i = blockIdx.x * blockDim.x + threadIdx.x;
  if (i >= m) return;
  const int s = idx[i];
  out[i * 3 + 0] = pos[s * 3 + 0];
  out[i * 3 + 1] = pos[s * 3 + 1];
  out[i * 3 + 2] = pos[s * 3 + 2];
}

__global__ void gather_max_kernel(const float* __restrict__ hh,
                                  const int* __restrict__ nbr,
                                  float* __restrict__ out, int m, int C, int Kn)
{
  int t = blockIdx.x * blockDim.x + threadIdx.x;
  if (t >= m * C) return;
  const int i = t / C, c = t - i * C;
  float mx = -3.4e38f;
  for (int j = 0; j < Kn; ++j)
    mx = fmaxf(mx, hh[(size_t)nbr[i * Kn + j] * C + c]);
  out[(size_t)i * C + c] = mx;
}

__global__ void mean_rows_kernel(const float* __restrict__ h,
                                 float* __restrict__ out, int n, int C)
{
  int c = blockIdx.x * blockDim.x + threadIdx.x;
  if (c >= C) return;
  float s = 0.0f;
  for (int i = 0; i < n; ++i) s += h[(size_t)i * C + c];
  out[c] = s / (float)n;
}

__global__ void concat_summit_kernel(const float* __restrict__ h,
                                     const float* __restrict__ xm,
                                     float* __restrict__ out, int n, int C)
{
  int t = blockIdx.x * blockDim.x + threadIdx.x;
  if (t >= n * 2 * C) return;
  const int i = t / (2 * C), c = t - i * 2 * C;
  out[t] = (c < C) ? h[(size_t)i * C + c] : xm[c - C];
}

__global__ void interp_add_kernel(const float* __restrict__ xsp,
                                  const int* __restrict__ nbr,
                                  const float* __restrict__ d2,
                                  float* __restrict__ h, int n, int C)
{
  int t = blockIdx.x * blockDim.x + threadIdx.x;
  if (t >= n * C) return;
  const int i = t / C, c = t - i * C;
  const float w0 = 1.0f / fmaxf(d2[i * 3 + 0], 1e-16f);
  const float w1 = 1.0f / fmaxf(d2[i * 3 + 1], 1e-16f);
  const float w2 = 1.0f / fmaxf(d2[i * 3 + 2], 1e-16f);
  const float ws = w0 + w1 + w2;
  const float xi = (w0 * xsp[(size_t)nbr[i * 3 + 0] * C + c] +
                    w1 * xsp[(size_t)nbr[i * 3 + 1] * C + c] +
                    w2 * xsp[(size_t)nbr[i * 3 + 2] * C + c]) / ws;
  h[(size_t)i * C + c] += xi;
}

__global__ void logsoftmax_kernel(const float* __restrict__ in,
                                  float* __restrict__ out, int n, int C)
{
  int i = blockIdx.x * blockDim.x + threadIdx.x;
  if (i >= n) return;
  const float* r = in + (size_t)i * C;
  float m = -3.4e38f;
  for (int c = 0; c < C; ++c) m = fmaxf(m, r[c]);
  float s = 0.0f;
  for (int c = 0; c < C; ++c) s += expf(r[c] - m);
  const float l = logf(s);
  float* o = out + (size_t)i * C;
  for (int c = 0; c < C; ++c) o[c] = r[c] - m - l;
}

// ------------------------- host-side plumbing -----------------------------
static inline int nblk(long long n, int t) { return (int)((n + t - 1) / t); }

template <int NCT>
static inline void gemm_dispatch_k(const float* A, const float* W,
                                   const float* bias, float* C,
                                   int M, int Kd, int N, int relu,
                                   dim3 grid, dim3 blk, hipStream_t s)
{
  if ((Kd & 31) == 0)
    gemm_wmma_kernel<NCT, true><<<grid, blk, 0, s>>>(A, W, bias, C, M, Kd, N, relu);
  else
    gemm_wmma_kernel<NCT, false><<<grid, blk, 0, s>>>(A, W, bias, C, M, Kd, N, relu);
}

static inline void launch_gemm(const float* A, const float* W, const float* bias,
                               float* C, int M, int Kd, int N, int relu,
                               hipStream_t s)
{
  dim3 grid((N + 31) / 32, (M + 15) / 16);
  dim3 blk(64);
  switch (N) {
    case 13:  gemm_dispatch_k<13>(A, W, bias, C, M, Kd, N, relu, grid, blk, s); break;
    case 32:  gemm_dispatch_k<32>(A, W, bias, C, M, Kd, N, relu, grid, blk, s); break;
    case 64:  gemm_dispatch_k<64>(A, W, bias, C, M, Kd, N, relu, grid, blk, s); break;
    case 128: gemm_dispatch_k<128>(A, W, bias, C, M, Kd, N, relu, grid, blk, s); break;
    case 256: gemm_dispatch_k<256>(A, W, bias, C, M, Kd, N, relu, grid, blk, s); break;
    case 512: gemm_dispatch_k<512>(A, W, bias, C, M, Kd, N, relu, grid, blk, s); break;
    default:  gemm_dispatch_k<0>(A, W, bias, C, M, Kd, N, relu, grid, blk, s); break;
  }
}

struct Bump { char* base; size_t off; };
static inline void* balloc(Bump& b, size_t bytes)
{
  b.off = (b.off + 255) & ~(size_t)255;
  void* p = b.base + b.off;
  b.off += bytes;
  return p;
}

struct TBP {
  const float *Win, *bin, *Wout, *bout, *Wsrc, *Wdst, *Wval;
  const float *P1, *pb1, *P2, *pb2, *A1, *ab1, *A2, *ab2;
};

static TBP parse_tb(void* const* d_in, int& t)
{
  TBP p;
  p.Win = (const float*)d_in[t++]; p.bin = (const float*)d_in[t++];
  p.Wout = (const float*)d_in[t++]; p.bout = (const float*)d_in[t++];
  p.Wsrc = (const float*)d_in[t++];
  p.Wdst = (const float*)d_in[t++];
  p.Wval = (const float*)d_in[t++];
  p.P1 = (const float*)d_in[t++]; p.pb1 = (const float*)d_in[t++];
  p.P2 = (const float*)d_in[t++]; p.pb2 = (const float*)d_in[t++];
  p.A1 = (const float*)d_in[t++]; p.ab1 = (const float*)d_in[t++];
  p.A2 = (const float*)d_in[t++]; p.ab2 = (const float*)d_in[t++];
  return p;
}

// One TransformerBlock. `b` is a by-value bump copy -> temps auto-recycle.
static void run_tb(const TBP& p, const float* hin, float* hout,
                   const float* pos, const int* src, const int* dst,
                   int n, int C, Bump b, hipStream_t stream)
{
  const int E = n * WKNN;
  float* x1    = (float*)balloc(b, (size_t)n * C * 4);
  float* asrc  = (float*)balloc(b, (size_t)n * C * 4);
  float* adst  = (float*)balloc(b, (size_t)n * C * 4);
  float* v     = (float*)balloc(b, (size_t)n * C * 4);
  float* tB    = (float*)balloc(b, (size_t)E * C * 4);   // pd / alpha_pre / alpha
  float* tA    = (float*)balloc(b, (size_t)E * 64 * 4);  // MLP hidden
  float* delta = (float*)balloc(b, (size_t)E * C * 4);

  launch_gemm(hin, p.Win, p.bin, x1, n, C, C, 1, stream);           // relu(lin_in)
  launch_gemm(x1, p.Wsrc, nullptr, asrc, n, C, C, 0, stream);
  launch_gemm(x1, p.Wdst, nullptr, adst, n, C, C, 0, stream);
  launch_gemm(x1, p.Wval, nullptr, v, n, C, C, 0, stream);

  edge_posdiff_kernel<<<nblk(E, 256), 256, 0, stream>>>(pos, src, dst, tB, E);
  launch_gemm(tB, p.P1, p.pb1, tA, E, 3, 64, 1, stream);            // pos_nn L1
  launch_gemm(tA, p.P2, p.pb2, delta, E, 64, C, 1, stream);         // pos_nn L2

  edge_alpha_pre_kernel<<<nblk((long long)E * C, 256), 256, 0, stream>>>(
      adst, asrc, delta, src, dst, tB, E, C);
  launch_gemm(tB, p.A1, p.ab1, tA, E, C, 64, 1, stream);            // attn_nn L1
  launch_gemm(tA, p.A2, p.ab2, tB, E, 64, C, 1, stream);            // attn_nn L2

  tb_aggregate_kernel<<<nblk((long long)n * C, 256), 256, 0, stream>>>(
      tB, v, delta, src, x1 /*reuse as out*/, n, C, WKNN);
  launch_gemm(x1, p.Wout, p.bout, hout, n, C, C, 1, stream);        // relu(lin_out)
}

extern "C" void kernel_launch(void* const* d_in, const int* in_sizes, int n_in,
                              void* d_out, int out_size, void* d_ws, size_t ws_size,
                              hipStream_t stream)
{
  (void)in_sizes; (void)n_in; (void)out_size; (void)ws_size;
  const int NS[5] = {8192, 2048, 512, 128, 32};
  const int CS[5] = {32, 64, 128, 256, 512};
  const int BL[4] = {2, 3, 5, 2};

  // ---- parse inputs in setup_inputs() dict order ----
  int t = 0;
  const float* x   = (const float*)d_in[t++];
  const float* pos = (const float*)d_in[t++];
  const float* Win_mlp = (const float*)d_in[t++];
  TBP tb_in = parse_tb(d_in, t);
  struct EncP { const float *W, *b; TBP tbs[5]; } enc[4];
  for (int i = 0; i < 4; ++i) {
    enc[i].W = (const float*)d_in[t++];
    enc[i].b = (const float*)d_in[t++];
    for (int j = 0; j < BL[i]; ++j) enc[i].tbs[j] = parse_tb(d_in, t);
  }
  const float* Ws = (const float*)d_in[t++];
  const float* bs = (const float*)d_in[t++];
  const float* Wc = (const float*)d_in[t++];
  const float* bc = (const float*)d_in[t++];
  TBP tb_summit = parse_tb(d_in, t);
  struct DecP { const float *Wsub, *bsub, *Wm, *bm; TBP tb; } dec[4];
  for (int i = 0; i < 4; ++i) {
    dec[i].Wsub = (const float*)d_in[t++];
    dec[i].bsub = (const float*)d_in[t++];
    dec[i].Wm   = (const float*)d_in[t++];
    dec[i].bm   = (const float*)d_in[t++];
    dec[i].tb   = parse_tb(d_in, t);
  }
  const float* Wo1 = (const float*)d_in[t++];
  const float* bo1 = (const float*)d_in[t++];
  const float* Wo2 = (const float*)d_in[t++];
  const float* bo2 = (const float*)d_in[t++];

  // ---- persistent workspace layout ----
  Bump ws{(char*)d_ws, 0};
  const float* P[5];
  P[0] = pos;
  for (int l = 1; l < 5; ++l) P[l] = (float*)balloc(ws, (size_t)NS[l] * 3 * 4);
  float* XS[5];
  for (int l = 0; l < 5; ++l) XS[l] = (float*)balloc(ws, (size_t)NS[l] * CS[l] * 4);
  int *ESRC[5], *EDST[5];
  for (int l = 0; l < 5; ++l) {
    ESRC[l] = (int*)balloc(ws, (size_t)NS[l] * WKNN * 4);
    EDST[l] = (int*)balloc(ws, (size_t)NS[l] * WKNN * 4);
  }
  int*   fpsidx  = (int*)balloc(ws, 2048 * 4);
  float* fpsdist = (float*)balloc(ws, 8192 * 4);
  float* G0 = (float*)balloc(ws, (size_t)8192 * 64 * 4);
  float* G1 = (float*)balloc(ws, (size_t)8192 * 64 * 4);
  const Bump tmp = ws;  // transient region starts here (safe to alias across
                        // stages: stream ordering serializes kernels)

  // ---- input MLP: h = cat(pos, x) @ Win_mlp ----
  {
    Bump b = tmp;
    float* cat6 = (float*)balloc(b, (size_t)8192 * 6 * 4);
    cat_pos_x_kernel<<<nblk(8192, 256), 256, 0, stream>>>(pos, x, cat6, 8192);
    launch_gemm(cat6, Win_mlp, nullptr, G0, 8192, 6, 32, 0, stream);
  }
  // level-0 knn graph + tb_in
  knn_kernel<WKNN><<<nblk(8192, 128), 128, 0, stream>>>(pos, 8192, pos, 8192,
                                                        ESRC[0], nullptr);
  fill_dst_kernel<<<nblk(8192 * WKNN, 256), 256, 0, stream>>>(EDST[0],
                                                              8192 * WKNN, WKNN);
  run_tb(tb_in, G0, XS[0], P[0], ESRC[0], EDST[0], 8192, 32, tmp, stream);
  const float* hcur = XS[0];

  // ---- encoder: TransitionDown + transformer blocks ----
  for (int i = 0; i < 4; ++i) {
    Bump b = tmp;
    const int nf = NS[i], nc = NS[i + 1], Ci = CS[i], Co = CS[i + 1];
    fps_kernel<<<1, 512, 0, stream>>>(P[i], nf, nc, fpsidx, fpsdist);
    gather_pos_kernel<<<nblk(nc, 256), 256, 0, stream>>>(P[i], fpsidx,
                                                         (float*)P[i + 1], nc);
    int* nbrTD = (int*)balloc(b, (size_t)nc * WKNN * 4);
    knn_kernel<WKNN><<<nblk(nc, 128), 128, 0, stream>>>(P[i], nf, P[i + 1], nc,
                                                        nbrTD, nullptr);
    float* hh = (float*)balloc(b, (size_t)nf * Co * 4);
    launch_gemm(hcur, enc[i].W, enc[i].b, hh, nf, Ci, Co, 1, stream);
    gather_max_kernel<<<nblk((long long)nc * Co, 256), 256, 0, stream>>>(
        hh, nbrTD, G0, nc, Co, WKNN);
    knn_kernel<WKNN><<<nblk(nc, 128), 128, 0, stream>>>(P[i + 1], nc, P[i + 1],
                                                        nc, ESRC[i + 1], nullptr);
    fill_dst_kernel<<<nblk(nc * WKNN, 256), 256, 0, stream>>>(EDST[i + 1],
                                                              nc * WKNN, WKNN);
    const float* cur = G0;
    for (int j = 0; j < BL[i]; ++j) {
      float* nxt = (j == BL[i] - 1) ? XS[i + 1] : ((cur == G0) ? G1 : G0);
      run_tb(enc[i].tbs[j], cur, nxt, P[i + 1], ESRC[i + 1], EDST[i + 1],
             nc, Co, b, stream);
      cur = nxt;
    }
    hcur = XS[i + 1];
  }

  // ---- summit: global mean pool + concat + tb ----
  {
    Bump b = tmp;
    float* mv  = (float*)balloc(b, 512 * 4);
    float* xmv = (float*)balloc(b, 512 * 4);
    mean_rows_kernel<<<nblk(512, 256), 256, 0, stream>>>(hcur, mv, 32, 512);
    launch_gemm(mv, Ws, bs, xmv, 1, 512, 512, 1, stream);
    float* cat2 = (float*)balloc(b, (size_t)32 * 1024 * 4);
    concat_summit_kernel<<<nblk(32 * 1024, 256), 256, 0, stream>>>(hcur, xmv,
                                                                   cat2, 32, 512);
    launch_gemm(cat2, Wc, bc, G0, 32, 1024, 512, 0, stream);
    run_tb(tb_summit, G0, G1, P[4], ESRC[4], EDST[4], 32, 512, b, stream);
  }
  float* hc = G1;
  float* hother = G0;

  // ---- decoder: TransitionUp + transformer blocks ----
  for (int i = 0; i < 4; ++i) {
    Bump b = tmp;
    const int lf = 3 - i, lc = 4 - i;
    const int nf = NS[lf], ncn = NS[lc], ci = CS[lc], co = CS[lf];
    float* xsp = (float*)balloc(b, (size_t)ncn * co * 4);
    launch_gemm(hc, dec[i].Wsub, dec[i].bsub, xsp, ncn, ci, co, 0, stream);
    int*   nbr3 = (int*)balloc(b, (size_t)nf * 3 * 4);
    float* nd2  = (float*)balloc(b, (size_t)nf * 3 * 4);
    knn_kernel<3><<<nblk(nf, 128), 128, 0, stream>>>(P[lc], ncn, P[lf], nf,
                                                     nbr3, nd2);
    float* hnew = hother;
    launch_gemm(XS[lf], dec[i].Wm, dec[i].bm, hnew, nf, co, co, 0, stream);
    interp_add_kernel<<<nblk((long long)nf * co, 256), 256, 0, stream>>>(
        xsp, nbr3, nd2, hnew, nf, co);
    float* hout = hc;  // old buffer already consumed by xsp GEMM (stream order)
    run_tb(dec[i].tb, hnew, hout, P[lf], ESRC[lf], EDST[lf], nf, co, b, stream);
    hc = hout; hother = hnew;
  }

  // ---- output head + log_softmax ----
  {
    Bump b = tmp;
    float* o1 = hother;
    launch_gemm(hc, Wo1, bo1, o1, 8192, 32, 32, 0, stream);
    float* olin = (float*)balloc(b, (size_t)8192 * 13 * 4);
    launch_gemm(o1, Wo2, bo2, olin, 8192, 32, 13, 0, stream);
    logsoftmax_kernel<<<nblk(8192, 256), 256, 0, stream>>>(olin, (float*)d_out,
                                                           8192, 13);
  }
}
